// PointPillars_70712341562126
// MI455X (gfx1250) — matrix operand order repaired and checked by reading, not compiled, not run on hardware
//
#include <hip/hip_runtime.h>

typedef __attribute__((ext_vector_type(16))) _Float16 v16h;
typedef __attribute__((ext_vector_type(8)))  _Float16 v8h;
typedef __attribute__((ext_vector_type(8)))  float    v8f;
typedef _Float16 h16;

#define EPSV 1e-5f
#define ASYNC_B 1  // stage B tile via global_load_async_to_lds_b128 (ASYNCcnt path)

static inline int cdiv(int a, int b) { return (a + b - 1) / b; }

__device__ inline unsigned lds_addr32(const void* p) {
  // generic LDS address: low 32 bits are the LDS byte offset (ISA aperture rule)
  return (unsigned)(unsigned long long)(uintptr_t)p;
}

// ---------------------------------------------------------------------------
// utility kernels
// ---------------------------------------------------------------------------
__global__ void k_fill_u32(unsigned* p, long n) {
  long i = (long)blockIdx.x * blockDim.x + threadIdx.x;
  if (i < n) p[i] = 0u;
}

__global__ void k_f32_to_h(const float* __restrict__ in, h16* __restrict__ out, long n) {
  long i = (long)blockIdx.x * blockDim.x + threadIdx.x;
  if (i < n) out[i] = (h16)in[i];
}

// fold conv bias + batchnorm into per-channel scale/shift:
//   y = conv*s + t,  s = g*rsqrt(v+eps),  t = bn_b + (bias - m)*s
__global__ void k_prep_ss(const float* g, const float* bb, const float* m,
                          const float* v, const float* bias,
                          float* sc, float* sh, int C, int has_bn) {
  int c = blockIdx.x * blockDim.x + threadIdx.x;
  if (c >= C) return;
  float b = bias ? bias[c] : 0.f;
  if (has_bn) {
    float s = g[c] * rsqrtf(v[c] + EPSV);
    sc[c] = s;
    sh[c] = bb[c] + (b - m[c]) * s;
  } else {
    sc[c] = 1.f;
    sh[c] = b;
  }
}

// pack OIHW f32 conv weights -> f16 K-major layout [Co64][K = Ci*ks*ks]
// (row per output channel: contiguous K makes B tiles a straight 16B/lane copy)
__global__ void k_pack_conv(const float* __restrict__ w, h16* __restrict__ out,
                            int Co, int Ci, int ks, int Co64) {
  int K = Ci * ks * ks;
  long total = (long)Co64 * K;
  long i = (long)blockIdx.x * blockDim.x + threadIdx.x;
  if (i >= total) return;
  int o = (int)(i / K);
  int k = (int)(i - (long)o * K);
  float val = 0.f;
  if (o < Co) {
    int kk = ks * ks;
    int ci = k / kk, rs = k - ci * kk;
    int r = rs / ks, s = rs - r * ks;
    val = w[(((long)o * Ci + ci) * ks + r) * ks + s];
  }
  out[i] = (h16)val;
}

// pack (Ci,Co,2,2) deconv weights -> f16 [4 (d,k)][Co64][Ci] (K-major)
__global__ void k_pack_deconv(const float* __restrict__ w, h16* __restrict__ out,
                              int Co, int Ci, int Co64) {
  long total = 4L * Co64 * Ci;
  long i = (long)blockIdx.x * blockDim.x + threadIdx.x;
  if (i >= total) return;
  int dk = (int)(i / ((long)Co64 * Ci));
  long rem = i - (long)dk * Co64 * Ci;
  int o = (int)(rem / Ci);
  int k = (int)(rem - (long)o * Ci);
  float val = 0.f;
  if (o < Co) val = w[(((long)k * Co + o) * 2 + (dk >> 1)) * 2 + (dk & 1)];
  out[i] = (h16)val;
}

// ---------------------------------------------------------------------------
// PFN: per-pillar MLP 9->64 (bn,relu), 64->64 (bn,relu), max over 32 points
// ---------------------------------------------------------------------------
__global__ __launch_bounds__(64) void k_pfn(
    const float* __restrict__ pts,
    const float* __restrict__ w1, const float* __restrict__ s1, const float* __restrict__ t1,
    const float* __restrict__ w2, const float* __restrict__ s2, const float* __restrict__ t2,
    float* __restrict__ feat) {
  int p = blockIdx.x, d = threadIdx.x;  // d = channel 0..63
  __shared__ float sp[32 * 9];
  __shared__ float sx[32 * 64];
  for (int i = d; i < 32 * 9; i += 64) sp[i] = pts[(long)p * 288 + i];
  __syncthreads();
  float wr[9];
#pragma unroll
  for (int c = 0; c < 9; ++c) wr[c] = w1[d * 9 + c];
  float sA = s1[d], tA = t1[d];
  for (int n = 0; n < 32; ++n) {
    float acc = 0.f;
#pragma unroll
    for (int c = 0; c < 9; ++c) acc += sp[n * 9 + c] * wr[c];
    sx[n * 64 + d] = fmaxf(acc * sA + tA, 0.f);
  }
  __syncthreads();
  float w2r[64];
#pragma unroll
  for (int k = 0; k < 64; ++k) w2r[k] = w2[d * 64 + k];
  float sB = s2[d], tB = t2[d], mx = 0.f;  // post-ReLU values are >= 0
  for (int n = 0; n < 32; ++n) {
    float acc = 0.f;
#pragma unroll
    for (int k = 0; k < 64; ++k) acc += sx[n * 64 + k] * w2r[k];
    mx = fmaxf(mx, fmaxf(acc * sB + tB, 0.f));
  }
  feat[(long)p * 64 + d] = mx;
}

// scatter-max into BEV grid. feat >= 0, so uint bit-pattern max == float max.
__global__ void k_scatter(const float* __restrict__ feat, const int* __restrict__ coords,
                          unsigned* __restrict__ bev, int HW, int Wd, int Hd, int P) {
  int i = blockIdx.x * blockDim.x + threadIdx.x;
  if (i >= P * 64) return;
  int p = i >> 6, c = i & 63;
  int xi = coords[p * 2], yi = coords[p * 2 + 1];
  if (xi < 0 || xi >= Wd || yi < 0 || yi >= Hd) return;
  atomicMax(&bev[(long)c * HW + yi * Wd + xi], __float_as_uint(feat[i]));
}

// ---------------------------------------------------------------------------
// Implicit-GEMM conv via v_wmma_f32_16x16x32_f16, wave32.
//   block = 256 threads (8 waves), block tile = 128 pixels x 64 out-channels,
//   wave tile = 32x32 (4 accumulators: 4 WMMAs per 8 ds_load_b128 per K-step).
//   K stepped by 32, LDS double-buffered (1 barrier/step), fragment-ready
//   16B-aligned layouts (A [pixel][k], B [ch][k], both K-contiguous).
//   B tile: weights are pre-packed K-major, so each thread copies one
//   contiguous 16B run -> staged with GLOBAL_LOAD_ASYNC_TO_LDS_B128
//   (ASYNCcnt, fenced with s_wait_asynccnt before the publishing barrier).
//   dk>=0: deconv mode (1x1 GEMM, output scattered to (2y+d, 2x+k)).
//   Epilogue: y = acc*scale[co]+shift[co]; optional ReLU; optional skip add.
// ---------------------------------------------------------------------------
__global__ __launch_bounds__(256) void k_conv_wmma(
    const h16* __restrict__ in, const h16* __restrict__ wB,
    const float* __restrict__ sc, const float* __restrict__ sh,
    const h16* __restrict__ skip, h16* __restrict__ out_h, float* __restrict__ out_f,
    int Ci, int Hi, int Wi, int Co, int Ho, int Wo,
    int ks, int stride, int pad, int relu, int dk) {
  const int K = Ci * ks * ks;  // always a multiple of 32 in this network
  const int M = Ho * Wo;
  const int m0 = blockIdx.x * 128;
  const int n0 = blockIdx.y * 64;
  const int tid = threadIdx.x;
  const int lane = tid & 31;
  const int wid = tid >> 5;
  const int wm = (wid & 3) << 5;   // wave M offset: 0,32,64,96
  const int wn = (wid >> 2) << 5;  // wave N offset: 0,32
  const int hsel = lane >> 4;      // ISA fragment lane-half select
  const int lr = lane & 15;

  // row stride 40 halves = 80B keeps every fragment run 16B-aligned
  __shared__ __align__(16) h16 sA[2][128][40];  // [pixel][k]
  __shared__ __align__(16) h16 sBt[2][64][40];  // [out-ch][k]

  // ---- hoisted im2col coords: each thread owns one output pixel ----
  const int miA = tid & 127;
  const int kgA = tid >> 7;  // k-subgroup 0..1
  const int mA = m0 + miA;
  int yA = 0, xA = 0;
  if (mA < M) { yA = mA / Wo; xA = mA - yA * Wo; }
  const int iy0 = yA * stride - pad;
  const int ix0 = xA * stride - pad;
  const long HiWi = (long)Hi * Wi;

  // ---- hoisted B staging coords: one 16B run per thread ----
  const int nrB = tid >> 2;        // out-channel row 0..63
  const int ksB = (tid & 3) << 3;  // k segment start: 0,8,16,24

  auto stage = [&](int buf, int kb) {
    // A tile: 16 boundary-checked scalar loads, fixed pixel, k = kgA + e*2
#pragma unroll
    for (int e = 0; e < 16; ++e) {
      int kk = kgA + e * 2;
      int k = kb + kk;
      float val = 0.f;
      if (mA < M) {
        int ci, r, s;
        if (ks == 3) { ci = k / 9; int rs = k - ci * 9; r = rs / 3; s = rs - r * 3; }
        else         { ci = k; r = 0; s = 0; }
        int iy = iy0 + r, ix = ix0 + s;
        if (iy >= 0 && iy < Hi && ix >= 0 && ix < Wi)
          val = (float)in[ci * HiWi + (long)iy * Wi + ix];
      }
      sA[buf][miA][kk] = (h16)val;
    }
    // B tile: contiguous 16B per lane (weights are K-major)
    const h16* gsrc = &wB[(long)(n0 + nrB) * K + kb + ksB];
    h16* ldst = &sBt[buf][nrB][ksB];
#if ASYNC_B
    asm volatile("global_load_async_to_lds_b128 %0, %1, off"
                 :
                 : "v"(lds_addr32(ldst)),
                   "v"((unsigned long long)(uintptr_t)gsrc)
                 : "memory");
#else
    *(v8h*)ldst = *(const v8h*)gsrc;
#endif
    if (kb + 64 < K)  // global_prefetch_b8 the slice after next
      __builtin_prefetch(&wB[(long)(n0 + nrB) * K + kb + 64], 0, 0);
  };

  v8f c00 = {}, c01 = {}, c10 = {}, c11 = {};

  stage(0, 0);
#if ASYNC_B
  asm volatile("s_wait_asynccnt 0x0" ::: "memory");
#endif
  __syncthreads();

  const int mrow0 = wm + lr;       // rows of A-frag 0
  const int mrow1 = wm + 16 + lr;  // rows of A-frag 1
  const int kbase = hsel << 3;  // A: lanes 0-15 hold K 0..7/16..23, 16-31 hold 8..15/24..31
  const int col0 = wn + lr;
  const int col1 = wn + 16 + lr;
  const int krow = hsel << 3;   // B: lanes 0-15 hold K 0..7/16..23, 16-31 hold 8..15/24..31

  for (int kb = 0; kb < K; kb += 32) {
    const int cur = (kb >> 5) & 1;
    if (kb + 32 < K) stage(cur ^ 1, kb + 32);  // overlap staging with WMMA

    // fragment assembly: eight 128-bit LDS loads per lane feed 4 WMMAs
    v8h a0lo = *(const v8h*)&sA[cur][mrow0][kbase];
    v8h a0hi = *(const v8h*)&sA[cur][mrow0][16 + kbase];
    v8h a1lo = *(const v8h*)&sA[cur][mrow1][kbase];
    v8h a1hi = *(const v8h*)&sA[cur][mrow1][16 + kbase];
    v8h b0lo = *(const v8h*)&sBt[cur][col0][krow];
    v8h b0hi = *(const v8h*)&sBt[cur][col0][krow + 16];
    v8h b1lo = *(const v8h*)&sBt[cur][col1][krow];
    v8h b1hi = *(const v8h*)&sBt[cur][col1][krow + 16];
    v16h a0, a1, b0, b1;
#pragma unroll
    for (int i = 0; i < 8; ++i) {
      a0[i] = a0lo[i]; a0[8 + i] = a0hi[i];
      a1[i] = a1lo[i]; a1[8 + i] = a1hi[i];
      b0[i] = b0lo[i]; b0[8 + i] = b0hi[i];
      b1[i] = b1lo[i]; b1[8 + i] = b1hi[i];
    }
    c00 = __builtin_amdgcn_wmma_f32_16x16x32_f16(false, a0, false, b0, (short)0, c00, false, false);
    c01 = __builtin_amdgcn_wmma_f32_16x16x32_f16(false, a0, false, b1, (short)0, c01, false, false);
    c10 = __builtin_amdgcn_wmma_f32_16x16x32_f16(false, a1, false, b0, (short)0, c10, false, false);
    c11 = __builtin_amdgcn_wmma_f32_16x16x32_f16(false, a1, false, b1, (short)0, c11, false, false);
#if ASYNC_B
    asm volatile("s_wait_asynccnt 0x0" ::: "memory");
#endif
    __syncthreads();
  }

  // ---- epilogue: C/D layout -> VGPR r holds M=r (lanes 0-15) / M=8+r ----
  const int co0 = n0 + wn + lr;
  const int co1 = co0 + 16;
  float s0 = 0.f, t0 = 0.f, s1 = 0.f, t1 = 0.f;
  if (co0 < Co) { s0 = sc[co0]; t0 = sh[co0]; }
  if (co1 < Co) { s1 = sc[co1]; t1 = sh[co1]; }

#pragma unroll
  for (int mt = 0; mt < 2; ++mt) {
#pragma unroll
    for (int r = 0; r < 8; ++r) {
      int m = m0 + wm + mt * 16 + (hsel << 3) + r;
      if (m >= M) continue;
      int y = 0, x = 0;
      if (dk >= 0) { y = m / Wo; x = m - y * Wo; }
#pragma unroll
      for (int nt = 0; nt < 2; ++nt) {
        int co = nt ? co1 : co0;
        if (co >= Co) continue;
        float acc = mt ? (nt ? c11[r] : c10[r]) : (nt ? c01[r] : c00[r]);
        float v = acc * (nt ? s1 : s0) + (nt ? t1 : t0);
        if (relu) v = fmaxf(v, 0.f);
        long oidx;
        if (dk >= 0) {  // deconv scatter: out dims (2Ho, 2Wo)
          oidx = (long)co * (4 * M) + (long)(2 * y + (dk >> 1)) * (2 * Wo) + (2 * x + (dk & 1));
        } else {
          oidx = (long)co * M + m;
        }
        if (skip) v += (float)skip[oidx];
        if (out_h) out_h[oidx] = (h16)v;
        if (out_f) out_f[oidx] = v;
      }
    }
  }
}

// head reshape: out[m*2*last + ch] = conv[ch][m]   (ch = a*last + c)
__global__ void k_head(const float* __restrict__ in, float* __restrict__ out, int Cout, int HW) {
  long i = (long)blockIdx.x * blockDim.x + threadIdx.x;
  if (i >= (long)HW * Cout) return;
  int m = (int)(i / Cout);
  int ch = (int)(i - (long)m * Cout);
  out[i] = in[(long)ch * HW + m];
}

// ---------------------------------------------------------------------------
// host orchestration
// ---------------------------------------------------------------------------
static void* wsa(char*& cur, size_t bytes) {
  void* p = (void*)cur;
  cur += (bytes + 255) & ~(size_t)255;
  return p;
}

extern "C" void kernel_launch(void* const* d_in, const int* in_sizes, int n_in,
                              void* d_out, int out_size, void* d_ws, size_t ws_size,
                              hipStream_t stream) {
  if (!d_in || n_in < 90 || !d_out || !d_ws) return;
  (void)in_sizes; (void)out_size; (void)ws_size;

  const int P = 12000;
  const int H = 496, W = 432, HW = H * W;
  const int H2 = 248, W2 = 216, HW2 = H2 * W2;
  const int H3 = 124, W3 = 108, HW3 = H3 * W3;
  (void)HW3;

  const float* pts = (const float*)d_in[0];
  const int* coords = (const int*)d_in[1];
  // d_in[2], d_in[3]: H, W (device scalars; shapes hardcoded for graph capture)

  int ii = 4;
#define NXT ((const float*)d_in[ii++])
  const float *pfn_w1 = NXT, *pfn_b1 = NXT, *pbn1_g = NXT, *pbn1_b = NXT, *pbn1_m = NXT, *pbn1_v = NXT;
  const float *pfn_w2 = NXT, *pfn_b2 = NXT, *pbn2_g = NXT, *pbn2_b = NXT, *pbn2_m = NXT, *pbn2_v = NXT;
  const float *b1c1_w = NXT, *b1c1_b = NXT, *b1n1_g = NXT, *b1n1_b = NXT, *b1n1_m = NXT, *b1n1_v = NXT;
  const float *b1c2_w = NXT, *b1c2_b = NXT, *b1n2_g = NXT, *b1n2_b = NXT, *b1n2_m = NXT, *b1n2_v = NXT;
  const float *b2c1_w = NXT, *b2c1_b = NXT, *b2n1_g = NXT, *b2n1_b = NXT, *b2n1_m = NXT, *b2n1_v = NXT;
  const float *b2c2_w = NXT, *b2c2_b = NXT, *b2n2_g = NXT, *b2n2_b = NXT, *b2n2_m = NXT, *b2n2_v = NXT;
  const float *b3c1_w = NXT, *b3c1_b = NXT, *b3n1_g = NXT, *b3n1_b = NXT, *b3n1_m = NXT, *b3n1_v = NXT;
  const float *b3c2_w = NXT, *b3c2_b = NXT, *b3n2_g = NXT, *b3n2_b = NXT, *b3n2_m = NXT, *b3n2_v = NXT;
  const float *u1_w = NXT, *u1_b = NXT, *u1n_g = NXT, *u1n_b = NXT, *u1n_m = NXT, *u1n_v = NXT;
  const float *u2_w = NXT, *u2_b = NXT, *u2n_g = NXT, *u2n_b = NXT, *u2n_m = NXT, *u2n_v = NXT;
  const float *fin_w = NXT, *fin_b = NXT;
  const float *clsc_w = NXT, *clsc_b = NXT, *clsn_g = NXT, *clsn_b = NXT, *clsn_m = NXT, *clsn_v = NXT;
  const float *cls2_w = NXT, *cls2_b = NXT;
  const float *regc_w = NXT, *regc_b = NXT, *regn_g = NXT, *regn_b = NXT, *regn_m = NXT, *regn_v = NXT;
  const float *reg2_w = NXT, *reg2_b = NXT;
  const float *dirc_w = NXT, *dirc_b = NXT, *dirn_g = NXT, *dirn_b = NXT, *dirn_m = NXT, *dirn_v = NXT;
  const float *dir2_w = NXT, *dir2_b = NXT;
#undef NXT

  char* cur = (char*)d_ws;

  auto prep = [&](const float* g, const float* b, const float* m, const float* v,
                  const float* bias, int C) -> float* {
    float* p = (float*)wsa(cur, sizeof(float) * 2 * C);
    k_prep_ss<<<cdiv(C, 64), 64, 0, stream>>>(g, b, m, v, bias, p, p + C, C, g != nullptr);
    return p;
  };
  auto pack = [&](const float* w, int Co, int Ci, int ks) -> h16* {
    int Co64 = (Co + 63) & ~63;
    long n = (long)Co64 * Ci * ks * ks;
    h16* p = (h16*)wsa(cur, sizeof(h16) * n);
    k_pack_conv<<<cdiv((int)n, 256), 256, 0, stream>>>(w, p, Co, Ci, ks, Co64);
    return p;
  };
  auto packd = [&](const float* w, int Ci, int Co) -> h16* {
    int Co64 = (Co + 63) & ~63;
    long n = 4L * Co64 * Ci;
    h16* p = (h16*)wsa(cur, sizeof(h16) * n);
    k_pack_deconv<<<cdiv((int)n, 256), 256, 0, stream>>>(w, p, Co, Ci, Co64);
    return p;
  };
  auto conv = [&](const h16* in, const h16* wB, const float* ss, const h16* skip,
                  h16* oh, float* of, int Ci, int Hi, int Wi, int Co, int Ho, int Wo,
                  int ks, int stride, int pad, int relu, int dk) {
    int M = Ho * Wo;
    int tilesM = cdiv(M, 128);
    int tilesN = ((Co + 63) & ~63) / 64;
    k_conv_wmma<<<dim3(tilesM, tilesN), 256, 0, stream>>>(
        in, wB, ss, ss + Co, skip, oh, of, Ci, Hi, Wi, Co, Ho, Wo,
        ks, stride, pad, relu, dk);
  };

  // ---- fold BN, pack weights (deterministic, recomputed every call) ----
  float* ss_p1 = prep(pbn1_g, pbn1_b, pbn1_m, pbn1_v, pfn_b1, 64);
  float* ss_p2 = prep(pbn2_g, pbn2_b, pbn2_m, pbn2_v, pfn_b2, 64);
  float* ss_b1c1 = prep(b1n1_g, b1n1_b, b1n1_m, b1n1_v, b1c1_b, 64);
  float* ss_b1c2 = prep(b1n2_g, b1n2_b, b1n2_m, b1n2_v, b1c2_b, 64);
  float* ss_b2c1 = prep(b2n1_g, b2n1_b, b2n1_m, b2n1_v, b2c1_b, 128);
  float* ss_b2c2 = prep(b2n2_g, b2n2_b, b2n2_m, b2n2_v, b2c2_b, 128);
  float* ss_b3c1 = prep(b3n1_g, b3n1_b, b3n1_m, b3n1_v, b3c1_b, 256);
  float* ss_b3c2 = prep(b3n2_g, b3n2_b, b3n2_m, b3n2_v, b3c2_b, 256);
  float* ss_u1 = prep(u1n_g, u1n_b, u1n_m, u1n_v, u1_b, 128);
  float* ss_u2 = prep(u2n_g, u2n_b, u2n_m, u2n_v, u2_b, 64);
  float* ss_fin = prep(nullptr, nullptr, nullptr, nullptr, fin_b, 256);
  float* ss_clsc = prep(clsn_g, clsn_b, clsn_m, clsn_v, clsc_b, 256);
  float* ss_regc = prep(regn_g, regn_b, regn_m, regn_v, regc_b, 256);
  float* ss_dirc = prep(dirn_g, dirn_b, dirn_m, dirn_v, dirc_b, 256);
  float* ss_cls2 = prep(nullptr, nullptr, nullptr, nullptr, cls2_b, 6);
  float* ss_reg2 = prep(nullptr, nullptr, nullptr, nullptr, reg2_b, 14);
  float* ss_dir2 = prep(nullptr, nullptr, nullptr, nullptr, dir2_b, 4);

  h16* W_b1c1 = pack(b1c1_w, 64, 64, 3);
  h16* W_b1c2 = pack(b1c2_w, 64, 64, 3);
  h16* W_b2c1 = pack(b2c1_w, 128, 64, 3);
  h16* W_b2c2 = pack(b2c2_w, 128, 128, 3);
  h16* W_b3c1 = pack(b3c1_w, 256, 128, 3);
  h16* W_b3c2 = pack(b3c2_w, 256, 256, 3);
  h16* W_u1 = packd(u1_w, 256, 128);  // [4][128][256]
  h16* W_u2 = packd(u2_w, 128, 64);   // [4][64][128]
  h16* W_fin = pack(fin_w, 256, 64, 3);
  h16* W_clsc = pack(clsc_w, 256, 256, 3);
  h16* W_regc = pack(regc_w, 256, 256, 3);
  h16* W_dirc = pack(dirc_w, 256, 256, 3);
  h16* W_cls2 = pack(cls2_w, 6, 256, 1);
  h16* W_reg2 = pack(reg2_w, 14, 256, 1);
  h16* W_dir2 = pack(dir2_w, 4, 256, 1);

  // ---- activation buffers ----
  float* feat = (float*)wsa(cur, sizeof(float) * (long)P * 64);
  unsigned* bev32 = (unsigned*)wsa(cur, 4L * 64 * HW);
  h16* bevh = (h16*)wsa(cur, 2L * 64 * HW);
  h16* x1a = (h16*)wsa(cur, 2L * 64 * HW);
  h16* x1 = (h16*)wsa(cur, 2L * 64 * HW);
  h16* x2a = (h16*)wsa(cur, 2L * 128 * HW2);
  h16* x2 = (h16*)wsa(cur, 2L * 128 * HW2);
  h16* x3a = (h16*)wsa(cur, 2L * 256 * HW3);
  h16* x3 = (h16*)wsa(cur, 2L * 256 * HW3);
  h16* u1b = (h16*)wsa(cur, 2L * 128 * HW2);
  h16* u2b = (h16*)wsa(cur, 2L * 64 * HW);
  h16* fb = (h16*)wsa(cur, 2L * 256 * HW);
  h16* yb = (h16*)wsa(cur, 2L * 256 * HW);
  float* hob = (float*)wsa(cur, 4L * 14 * HW);

  // ---- PFN + scatter-max BEV ----
  k_pfn<<<P, 64, 0, stream>>>(pts, pfn_w1, ss_p1, ss_p1 + 64, pfn_w2, ss_p2, ss_p2 + 64, feat);
  long nbev = 64L * HW;
  k_fill_u32<<<cdiv((int)nbev, 256), 256, 0, stream>>>(bev32, nbev);
  k_scatter<<<cdiv(P * 64, 256), 256, 0, stream>>>(feat, coords, bev32, HW, W, H, P);
  k_f32_to_h<<<cdiv((int)nbev, 256), 256, 0, stream>>>((const float*)bev32, bevh, nbev);

  // ---- backbone ----
  conv(bevh, W_b1c1, ss_b1c1, nullptr, x1a, nullptr, 64, H, W, 64, H, W, 3, 1, 1, 1, -1);
  conv(x1a, W_b1c2, ss_b1c2, nullptr, x1, nullptr, 64, H, W, 64, H, W, 3, 1, 1, 1, -1);
  conv(x1, W_b2c1, ss_b2c1, nullptr, x2a, nullptr, 64, H, W, 128, H2, W2, 3, 2, 1, 1, -1);
  conv(x2a, W_b2c2, ss_b2c2, nullptr, x2, nullptr, 128, H2, W2, 128, H2, W2, 3, 1, 1, 1, -1);
  conv(x2, W_b3c1, ss_b3c1, nullptr, x3a, nullptr, 128, H2, W2, 256, H3, W3, 3, 2, 1, 1, -1);
  conv(x3a, W_b3c2, ss_b3c2, nullptr, x3, nullptr, 256, H3, W3, 256, H3, W3, 3, 1, 1, 1, -1);

  // deconv 256->128 (2x up) + relu(bn) + x2 skip, as 4 GEMMs
  for (int dk = 0; dk < 4; ++dk)
    conv(x3, W_u1 + (long)dk * 128 * 256, ss_u1, x2, u1b, nullptr,
         256, H3, W3, 128, H3, W3, 1, 1, 0, 1, dk);
  // deconv 128->64 (2x up) + relu(bn) + x1 skip
  for (int dk = 0; dk < 4; ++dk)
    conv(u1b, W_u2 + (long)dk * 64 * 128, ss_u2, x1, u2b, nullptr,
         128, H2, W2, 64, H2, W2, 1, 1, 0, 1, dk);

  // final conv 64->256, bias only, no relu
  conv(u2b, W_fin, ss_fin, nullptr, fb, nullptr, 64, H, W, 256, H, W, 3, 1, 1, 0, -1);

  // ---- heads ----
  float* out = (float*)d_out;
  // cls
  conv(fb, W_clsc, ss_clsc, nullptr, yb, nullptr, 256, H, W, 256, H, W, 3, 1, 1, 1, -1);
  conv(yb, W_cls2, ss_cls2, nullptr, nullptr, hob, 256, H, W, 6, H, W, 1, 1, 0, 0, -1);
  k_head<<<cdiv(HW * 6, 256), 256, 0, stream>>>(hob, out, 6, HW);
  // reg
  conv(fb, W_regc, ss_regc, nullptr, yb, nullptr, 256, H, W, 256, H, W, 3, 1, 1, 1, -1);
  conv(yb, W_reg2, ss_reg2, nullptr, nullptr, hob, 256, H, W, 14, H, W, 1, 1, 0, 0, -1);
  k_head<<<cdiv(HW * 14, 256), 256, 0, stream>>>(hob, out + (long)HW * 6, 14, HW);
  // dir
  conv(fb, W_dirc, ss_dirc, nullptr, yb, nullptr, 256, H, W, 256, H, W, 3, 1, 1, 1, -1);
  conv(yb, W_dir2, ss_dir2, nullptr, nullptr, hob, 256, H, W, 4, H, W, 1, 1, 0, 0, -1);
  k_head<<<cdiv(HW * 4, 256), 256, 0, stream>>>(hob, out + (long)HW * 20, 4, HW);
}
